// MolecularEncoder_78855599555025
// MI455X (gfx1250) — matrix operand order repaired and checked by reading, not compiled, not run on hardware
//
#include <hip/hip_runtime.h>
#include <hip/hip_bf16.h>

typedef __attribute__((ext_vector_type(16))) __bf16 v16bf;
typedef __attribute__((ext_vector_type(8)))  __bf16 v8bf;
typedef __attribute__((ext_vector_type(8)))  float  v8f;

#define NN   150000
#define NE   600000
#define NG   5000
#define FIN  6
#define HH   128
#define EO   256
#define LL   4
#define BN_EPS 1e-5f

// ---------------- elementwise / utility kernels ----------------

__global__ __launch_bounds__(256) void k_zero(float* __restrict__ p, int n) {
    int t = blockIdx.x * 256 + threadIdx.x;
    if (t < n) p[t] = 0.0f;
}

// h[n,c] = sum_f x[n,f] * Wn[f,c] + bn_b[c]
__global__ __launch_bounds__(256) void k_encoder(const float* __restrict__ x,
                                                 const float* __restrict__ Wn,
                                                 const float* __restrict__ bnb,
                                                 float* __restrict__ h) {
    int t = blockIdx.x * 256 + threadIdx.x;
    if (t >= NN * HH) return;
    int n = t >> 7, c = t & 127;
    float acc = bnb[c];
#pragma unroll
    for (int f = 0; f < FIN; ++f)
        acc = fmaf(x[n * FIN + f], Wn[f * HH + c], acc);
    h[t] = acc;
}

// scatter-add over edges: agg[dst] += h[src], 4 channels per thread
__global__ __launch_bounds__(256) void k_scatter(const float* __restrict__ h,
                                                 const int* __restrict__ ei,
                                                 float* __restrict__ agg) {
    int t = blockIdx.x * 256 + threadIdx.x;
    int e = t >> 5;
    if (e >= NE) return;
    int cg = (t & 31) << 2;
    int s = ei[e];
    int d = ei[NE + e];
    const float4 v = *(const float4*)(h + (size_t)s * HH + cg);
    float* p = agg + (size_t)d * HH + cg;
    atomicAdd(p + 0, v.x);
    atomicAdd(p + 1, v.y);
    atomicAdd(p + 2, v.z);
    atomicAdd(p + 3, v.w);
}

// m0 = bf16((1+eps)*h + agg)
__global__ __launch_bounds__(256) void k_m0(const float* __restrict__ h,
                                            const float* __restrict__ agg,
                                            const float* __restrict__ eps,
                                            int layer,
                                            __bf16* __restrict__ m0) {
    int t = blockIdx.x * 256 + threadIdx.x;
    if (t >= NN * HH) return;
    float e = eps[layer];
    m0[t] = (__bf16)((1.0f + e) * h[t] + agg[t]);
}

// repack f32 weight [128 x (ntTotal*16)] row-major into WMMA-B bf16 layout:
// Bp[((kb*ntTotal + nt)*32 + lane)*16 + i] = W[kb*32 + (lane<16?0:8) + (i&7) + (i>=8?16:0)][nt*16 + lane%16]
__global__ __launch_bounds__(256) void k_repack(const float* __restrict__ W,
                                                __bf16* __restrict__ Bp,
                                                int ntTotal) {
    int t = blockIdx.x * 256 + threadIdx.x;
    int total = 4 * ntTotal * 32 * 16;
    if (t >= total) return;
    int i    = t & 15;
    int lane = (t >> 4) & 31;
    int nt   = (t >> 9) % ntTotal;
    int kb   = (t >> 9) / ntTotal;
    int col  = nt * 16 + (lane & 15);
    int k    = kb * 32 + ((lane >> 4) << 3) + (i & 7) + ((i >> 3) << 4);
    Bp[t] = (__bf16)W[(size_t)k * (ntTotal * 16) + col];
}

// ---------------- WMMA GEMM: [rows x 128] (bf16) @ packed-B -> 8 col tiles ----------------
// 32 rows (two 16-row M-tiles) per wave: each B fragment feeds 2 WMMAs.
// RELU_BF16 = 1: out = bf16(relu(acc+bias)); 0: out = f32(acc+bias)
// NTT = total column tiles in the packed B matrix (compile-time for immediate offsets).
template<int RELU_BF16, int NTT>
__global__ __launch_bounds__(256) void k_gemm(const __bf16* __restrict__ A,
                                              const __bf16* __restrict__ Bp,
                                              const float* __restrict__ bias,
                                              void* __restrict__ Cout,
                                              int rows, int ldc, int ntBase) {
    const int lane     = threadIdx.x & 31;
    const int wave     = threadIdx.x >> 5;
    const int mTile    = blockIdx.x * 8 + wave;   // 32-row tile index
    const int row0     = mTile * 32;
    const int laneMod  = lane & 15;
    const int laneHalf = lane >> 4;

    const int aRow0 = row0 + laneMod;        // upper 16-row tile
    const int aRow1 = row0 + 16 + laneMod;   // lower 16-row tile
    const bool v0 = (aRow0 < rows);
    const bool v1 = (aRow1 < rows);

    // hoisted lane-base pointers; all fragment offsets below are compile-time
    const __bf16* aBase0 = A + (size_t)aRow0 * HH + laneHalf * 8;
    const __bf16* aBase1 = A + (size_t)aRow1 * HH + laneHalf * 8;
    const __bf16* bLane  = Bp + (size_t)(ntBase * 512 + lane * 16);

    v8f acc0[8] = {};
    v8f acc1[8] = {};

#pragma unroll
    for (int kb = 0; kb < 4; ++kb) {
        v16bf a0, a1;
        if (v0) {
            v8bf lo = *(const v8bf*)(aBase0 + kb * 32);
            v8bf hi = *(const v8bf*)(aBase0 + kb * 32 + 16);
#pragma unroll
            for (int i = 0; i < 8; ++i) { a0[i] = lo[i]; a0[i + 8] = hi[i]; }
        } else {
#pragma unroll
            for (int i = 0; i < 16; ++i) a0[i] = (__bf16)0.0f;
        }
        if (v1) {
            v8bf lo = *(const v8bf*)(aBase1 + kb * 32);
            v8bf hi = *(const v8bf*)(aBase1 + kb * 32 + 16);
#pragma unroll
            for (int i = 0; i < 8; ++i) { a1[i] = lo[i]; a1[i + 8] = hi[i]; }
        } else {
#pragma unroll
            for (int i = 0; i < 16; ++i) a1[i] = (__bf16)0.0f;
        }
#pragma unroll
        for (int nt = 0; nt < 8; ++nt) {
            v16bf b = *(const v16bf*)(bLane + (kb * NTT + nt) * 512);
            acc0[nt] = __builtin_amdgcn_wmma_f32_16x16x32_bf16(
                false, a0, false, b, (short)0, acc0[nt], false, false);
            acc1[nt] = __builtin_amdgcn_wmma_f32_16x16x32_bf16(
                false, a1, false, b, (short)0, acc1[nt], false, false);
        }
    }

#pragma unroll
    for (int nt = 0; nt < 8; ++nt) {
        int col = (ntBase + nt) * 16 + laneMod;
        float bv = bias[col];
#pragma unroll
        for (int r = 0; r < 8; ++r) {
            int rowU = row0 + r + laneHalf * 8;
            int rowL = rowU + 16;
            if (rowU < rows) {
                float v = acc0[nt][r] + bv;
                if (RELU_BF16) {
                    v = v > 0.0f ? v : 0.0f;
                    ((__bf16*)Cout)[(size_t)rowU * ldc + col] = (__bf16)v;
                } else {
                    ((float*)Cout)[(size_t)rowU * ldc + col] = v;
                }
            }
            if (rowL < rows) {
                float v = acc1[nt][r] + bv;
                if (RELU_BF16) {
                    v = v > 0.0f ? v : 0.0f;
                    ((__bf16*)Cout)[(size_t)rowL * ldc + col] = (__bf16)v;
                } else {
                    ((float*)Cout)[(size_t)rowL * ldc + col] = v;
                }
            }
        }
    }
}

// ---------------- batch norm ----------------

#define BN_ROWS_PER_BLOCK 256
__global__ __launch_bounds__(128) void k_bnstats(const float* __restrict__ m,
                                                 float* __restrict__ sums) {
    int c = threadIdx.x;              // 0..127
    size_t r0 = (size_t)blockIdx.x * BN_ROWS_PER_BLOCK;
    float s = 0.0f, s2 = 0.0f;
    for (int i = 0; i < BN_ROWS_PER_BLOCK; ++i) {
        size_t r = r0 + i;
        if (r < NN) {
            float v = m[r * HH + c];
            s += v;
            s2 = fmaf(v, v, s2);
        }
    }
    atomicAdd(&sums[c], s);
    atomicAdd(&sums[HH + c], s2);
}

// h = relu(gamma*(m-mu)*rsqrt(var+eps)+beta) + h
__global__ __launch_bounds__(256) void k_bnapply(const float* __restrict__ m,
                                                 const float* __restrict__ sums,
                                                 const float* __restrict__ gamma,
                                                 const float* __restrict__ beta,
                                                 float* __restrict__ h) {
    int t = blockIdx.x * 256 + threadIdx.x;
    if (t >= NN * HH) return;
    int c = t & 127;
    const float invN = 1.0f / (float)NN;
    float mu  = sums[c] * invN;
    float var = sums[HH + c] * invN - mu * mu;
    float v = gamma[c] * (m[t] - mu) * rsqrtf(var + BN_EPS) + beta[c];
    v = v > 0.0f ? v : 0.0f;
    h[t] = v + h[t];
}

// ---------------- pooling ----------------

__global__ __launch_bounds__(256) void k_pool(const float* __restrict__ h,
                                              const int* __restrict__ batch,
                                              float* __restrict__ psum,
                                              float* __restrict__ pcnt) {
    int t = blockIdx.x * 256 + threadIdx.x;
    if (t >= NN * HH) return;
    int n = t >> 7, c = t & 127;
    int g = batch[n];
    atomicAdd(&psum[(size_t)g * HH + c], h[t]);
    if (c == 0) atomicAdd(&pcnt[g], 1.0f);
}

#define NG_PAD 5024  // 5000 rounded up to multiple of 32
__global__ __launch_bounds__(256) void k_pooldiv(const float* __restrict__ psum,
                                                 const float* __restrict__ pcnt,
                                                 __bf16* __restrict__ pooled) {
    int t = blockIdx.x * 256 + threadIdx.x;
    if (t >= NG_PAD * HH) return;
    int g = t >> 7;
    float v = 0.0f;
    if (g < NG) {
        float cnt = pcnt[g];
        v = psum[t] / fmaxf(cnt, 1.0f);
    }
    pooled[t] = (__bf16)v;
}

// ---------------- host-side orchestration ----------------

static inline size_t alignUp(size_t v, size_t a) { return (v + a - 1) & ~(a - 1); }

extern "C" void kernel_launch(void* const* d_in, const int* in_sizes, int n_in,
                              void* d_out, int out_size, void* d_ws, size_t ws_size,
                              hipStream_t stream) {
    (void)in_sizes; (void)n_in; (void)out_size; (void)ws_size;

    const float* x     = (const float*)d_in[0];
    const int*   ei    = (const int*)d_in[1];
    const int*   batch = (const int*)d_in[2];
    const float* Wn    = (const float*)d_in[3];
    const float* bnb   = (const float*)d_in[4];
    const float* eps   = (const float*)d_in[5];
    const float* W1    = (const float*)d_in[6];
    const float* b1    = (const float*)d_in[7];
    const float* W2    = (const float*)d_in[8];
    const float* b2    = (const float*)d_in[9];
    const float* gamma = (const float*)d_in[10];
    const float* beta  = (const float*)d_in[11];
    const float* Wo1   = (const float*)d_in[12];
    const float* bo1   = (const float*)d_in[13];
    const float* Wo2   = (const float*)d_in[14];
    const float* bo2   = (const float*)d_in[15];
    float* out = (float*)d_out;

    // workspace carve-out
    char* ws = (char*)d_ws;
    size_t off = 0;
    auto take = [&](size_t bytes) { size_t o = off; off = alignUp(off + bytes, 256); return ws + o; };

    float*  h        = (float*)take((size_t)NN * HH * 4);
    float*  aggm     = (float*)take((size_t)NN * HH * 4);   // agg, then reused as GEMM2 f32 output
    __bf16* m0bf     = (__bf16*)take((size_t)NN * HH * 2);
    __bf16* ybf      = (__bf16*)take((size_t)NN * HH * 2);
    float*  sums     = (float*)take(2 * HH * 4);
    float*  psum     = (float*)take((size_t)NG * HH * 4);
    float*  pcnt     = (float*)take((size_t)NG * 4);
    __bf16* pooledbf = (__bf16*)take((size_t)NG_PAD * HH * 2);
    __bf16* yobf     = (__bf16*)take((size_t)NG_PAD * HH * 2);
    const size_t PK128 = (size_t)HH * HH;        // elements in packed 128x128
    __bf16* w1p  = (__bf16*)take(LL * PK128 * 2);
    __bf16* w2p  = (__bf16*)take(LL * PK128 * 2);
    __bf16* wo1p = (__bf16*)take(PK128 * 2);
    __bf16* wo2p = (__bf16*)take((size_t)HH * EO * 2);

    const int NH = NN * HH;
    const int gNH = (NH + 255) / 256;

    // repack all weights into WMMA-B layout (bf16)
    for (int i = 0; i < LL; ++i) {
        k_repack<<<(HH * HH + 255) / 256, 256, 0, stream>>>(W1 + (size_t)i * HH * HH, w1p + i * PK128, 8);
        k_repack<<<(HH * HH + 255) / 256, 256, 0, stream>>>(W2 + (size_t)i * HH * HH, w2p + i * PK128, 8);
    }
    k_repack<<<(HH * HH + 255) / 256, 256, 0, stream>>>(Wo1, wo1p, 8);
    k_repack<<<(HH * EO + 255) / 256, 256, 0, stream>>>(Wo2, wo2p, 16);

    // node encoder
    k_encoder<<<gNH, 256, 0, stream>>>(x, Wn, bnb, h);

    const int mTiles = (NN + 31) / 32;             // 4688
    const int gGemm = (mTiles + 7) / 8;            // 586

    for (int i = 0; i < LL; ++i) {
        // aggregate neighbors
        k_zero<<<gNH, 256, 0, stream>>>(aggm, NH);
        k_scatter<<<(NE * 32 + 255) / 256, 256, 0, stream>>>(h, ei, aggm);
        // m0 = bf16((1+eps)*h + agg)
        k_m0<<<gNH, 256, 0, stream>>>(h, aggm, eps, i, m0bf);
        // y = relu(m0 @ W1 + b1)   (bf16 out)
        k_gemm<1, 8><<<gGemm, 256, 0, stream>>>(m0bf, w1p + i * PK128, b1 + i * HH, ybf, NN, HH, 0);
        // m = y @ W2 + b2          (f32 out, reuse agg buffer)
        k_gemm<0, 8><<<gGemm, 256, 0, stream>>>(ybf, w2p + i * PK128, b2 + i * HH, aggm, NN, HH, 0);
        // batch norm stats + apply (+relu +residual into h)
        k_zero<<<1, 256, 0, stream>>>(sums, 2 * HH);
        k_bnstats<<<(NN + BN_ROWS_PER_BLOCK - 1) / BN_ROWS_PER_BLOCK, 128, 0, stream>>>(aggm, sums);
        k_bnapply<<<gNH, 256, 0, stream>>>(aggm, sums, gamma + i * HH, beta + i * HH, h);
    }

    // global mean pool
    k_zero<<<(NG * HH + 255) / 256, 256, 0, stream>>>(psum, NG * HH);
    k_zero<<<(NG + 255) / 256, 256, 0, stream>>>(pcnt, NG);
    k_pool<<<gNH, 256, 0, stream>>>(h, batch, psum, pcnt);
    k_pooldiv<<<(NG_PAD * HH + 255) / 256, 256, 0, stream>>>(psum, pcnt, pooledbf);

    // output head
    const int hTiles = (NG + 31) / 32;             // 157
    const int gHead = (hTiles + 7) / 8;            // 20
    k_gemm<1, 8><<<gHead, 256, 0, stream>>>(pooledbf, wo1p, bo1, yobf, NG, HH, 0);
    k_gemm<0, 16><<<gHead, 256, 0, stream>>>(yobf, wo2p, bo2, out, NG, EO, 0);
    k_gemm<0, 16><<<gHead, 256, 0, stream>>>(yobf, wo2p, bo2, out, NG, EO, 8);
}
